// GraphTransformer_6734508720201
// MI455X (gfx1250) — matrix operand order repaired and checked by reading, not compiled, not run on hardware
//
#include <hip/hip_runtime.h>
#include <hip/hip_bf16.h>
#include <math.h>

// Problem constants (match reference)
#define NODES 40000
#define EDGES 320000
#define BGR   8
#define NPG_  5000

typedef __attribute__((ext_vector_type(16))) _Float16 v16h;
typedef __attribute__((ext_vector_type(8)))  _Float16 v8h;
typedef __attribute__((ext_vector_type(8)))  float    v8f;
typedef __attribute__((ext_vector_type(4)))  unsigned int v4u;
typedef __attribute__((ext_vector_type(8)))  int      v8i_;
typedef __attribute__((ext_vector_type(4)))  int      v4i_;

// Tensor Data Mover available? (device pass only; host pass falls back)
#if defined(__gfx1250__) && __has_builtin(__builtin_amdgcn_tensor_load_to_lds) && \
    __has_builtin(__builtin_amdgcn_s_wait_tensorcnt)
#define USE_TDM 1
#else
#define USE_TDM 0
#endif

#if USE_TDM
// 2-D TDM load: tile = 128 rows x kc_halfs f16 elements, row stride k_stride_halfs,
// LDS rows padded by 4 DWORDs (8 halfs) -> matches KP = KC+8 layout.
// padc: pad_interval code (5 -> every 64 DWORDs, 3 -> every 16 DWORDs).
__device__ __forceinline__ void tdm_load_tile(unsigned ldsaddr, const void* gptr,
                                              int kc_halfs, int k_stride_halfs, int padc)
{
    unsigned long long ga = (unsigned long long)gptr;
    v4u g0 = { 1u,                                   // count=1, user descriptor
               ldsaddr,                              // lds_addr (bytes)
               (unsigned)(ga & 0xFFFFFFFFu),         // global_addr lo
               (unsigned)((ga >> 32) & 0x1FFFFFFu) | 0x80000000u };  // hi | type=2
    int d0 = (1 << 16)      // data_size = 2 bytes
           | (1 << 20)      // pad_enable
           | (padc << 22)   // pad_interval
           | (3 << 25);     // pad_amount = 4 DWORDs
    v8i_ g1 = { d0,
                (int)((kc_halfs & 0xFFFF) << 16),    // tensor_dim0 lo16
                (int)(128 << 16),                    // tensor_dim0 hi16 (0) | tensor_dim1 lo16
                (int)(kc_halfs << 16),               // tensor_dim1 hi16 (0) | tile_dim0
                128,                                 // tile_dim1 | tile_dim2(0)
                k_stride_halfs,                      // tensor_dim0_stride lo32
                0, 0 };
    v4i_ gz = { 0, 0, 0, 0 };
#if __clang_major__ >= 23
    v8i_ gz8 = { 0, 0, 0, 0, 0, 0, 0, 0 };
    __builtin_amdgcn_tensor_load_to_lds(g0, g1, gz, gz, gz8, 0);
#else
    __builtin_amdgcn_tensor_load_to_lds(g0, g1, gz, gz, 0);
#endif
}
#endif

// A fragment (16x32 f16): lane<16 holds K {0..7,16..23}, lane>=16 {8..15,24..31};
// p = &Arow[kb + half8] (16B aligned)
__device__ __forceinline__ v16h load_afrag(const float* __restrict__ p, bool valid)
{
    v16h a;
    if (valid) {
        float4 x0 = *(const float4*)(p);
        float4 x1 = *(const float4*)(p + 4);
        float4 y0 = *(const float4*)(p + 16);
        float4 y1 = *(const float4*)(p + 20);
        a[0]  = (_Float16)x0.x; a[1]  = (_Float16)x0.y;
        a[2]  = (_Float16)x0.z; a[3]  = (_Float16)x0.w;
        a[4]  = (_Float16)x1.x; a[5]  = (_Float16)x1.y;
        a[6]  = (_Float16)x1.z; a[7]  = (_Float16)x1.w;
        a[8]  = (_Float16)y0.x; a[9]  = (_Float16)y0.y;
        a[10] = (_Float16)y0.z; a[11] = (_Float16)y0.w;
        a[12] = (_Float16)y1.x; a[13] = (_Float16)y1.y;
        a[14] = (_Float16)y1.z; a[15] = (_Float16)y1.w;
    } else {
#pragma unroll
        for (int i = 0; i < 16; ++i) a[i] = (_Float16)0.0f;
    }
    return a;
}

// ---------------------------------------------------------------------------
// Fused GEMM: out[rows,128] = epilogue(A[rows,K] @ W[K,128] + bias)
// Wt = weights pre-transposed to [128 cols][K] f16 in global memory.
// MODE 0: (+bias) store   MODE 1: (+bias) relu   MODE 2: (+bias+res) LayerNorm
// Block = 256 threads = 8 waves; each wave owns TWO 16x128 output tiles so each
// B fragment (2x ds_load_b128) feeds two v_wmma_f32_16x16x32_f16.
// Weight tile staged in LDS by the Tensor Data Mover (pad -> KP=KC+8 layout).
// ---------------------------------------------------------------------------
template<int K, int MODE>
__global__ __launch_bounds__(256)
void gemm_fused(const float* __restrict__ A, int lda,
                const _Float16* __restrict__ Wt,
                const float* __restrict__ bias,
                const float* __restrict__ res,
                const float* __restrict__ gamma,
                const float* __restrict__ beta,
                float* __restrict__ out, int ldo, int coloff,
                int rows)
{
    constexpr int KC   = (K < 128) ? K : 128;   // K chunk held in LDS
    constexpr int KP   = KC + 8;                // +16B pad: b128-aligned, bank-spread
    constexpr int NCH  = K / KC;
    constexpr int PADC = (KC == 128) ? 5 : 3;   // TDM pad interval code (64/16 DWORDs)
    constexpr int KCS  = (KC == 128) ? 7 : 5;   // log2(KC)

    __shared__ __align__(16) _Float16 wlds[128 * KP];
    __shared__ float bias_s[128];
    __shared__ float g_s[128];
    __shared__ float b_s[128];

    const int tid     = threadIdx.x;
    const int lane    = tid & 31;
    const int wv      = tid >> 5;
    const int rowbase = blockIdx.x * 256 + wv * 32;   // 2 tiles of 16 rows per wave
    const int c0      = lane & 15;
    const int half8   = (lane >> 4) << 3;             // 0 or 8
    const int rowA0   = rowbase + c0;
    const int rowA1   = rowA0 + 16;
    const bool av0    = rowA0 < rows;
    const bool av1    = rowA1 < rows;

    if (tid < 128) {
        bias_s[tid] = bias ? bias[tid] : 0.0f;
        if (MODE == 2) { g_s[tid] = gamma[tid]; b_s[tid] = beta[tid]; }
    }

    v8f acc[2][8];
#pragma unroll
    for (int t = 0; t < 2; ++t)
#pragma unroll
        for (int cb = 0; cb < 8; ++cb)
#pragma unroll
            for (int vvi = 0; vvi < 8; ++vvi) acc[t][cb][vvi] = 0.0f;

    const float* Ar0 = A + (size_t)rowA0 * lda;
    const float* Ar1 = A + (size_t)rowA1 * lda;

    for (int ch = 0; ch < NCH; ++ch) {
        __syncthreads();   // previous chunk fully consumed before overwrite
#if USE_TDM
        if (wv == 0) {
            tdm_load_tile((unsigned)(unsigned long long)(uintptr_t)&wlds[0],
                          Wt + (size_t)ch * KC, KC, K, PADC);
            __builtin_amdgcn_s_wait_tensorcnt(0);
        }
#else
        for (int idx = tid; idx < 128 * KC; idx += 256) {
            int col = idx >> KCS, kk = idx & (KC - 1);
            wlds[col * KP + kk] = Wt[(size_t)col * K + ch * KC + kk];
        }
#endif
        __syncthreads();

#pragma unroll
        for (int kc = 0; kc < KC / 32; ++kc) {
            const int kb = ch * KC + kc * 32;
            v16h a0 = load_afrag(Ar0 + kb + half8, av0);
            v16h a1 = load_afrag(Ar1 + kb + half8, av1);
#pragma unroll
            for (int cb = 0; cb < 8; ++cb) {
                // B fragment (32x16 f16): col = lane%16, K = half8 + {0..7, 16..23}
                const v8h* bp = (const v8h*)
                    &wlds[(cb * 16 + c0) * KP + kc * 32 + half8];
                v8h blo = bp[0];
                v8h bhi = bp[2];
                v16h bfrag = __builtin_shufflevector(blo, bhi,
                    0, 1, 2, 3, 4, 5, 6, 7, 8, 9, 10, 11, 12, 13, 14, 15);
                acc[0][cb] = __builtin_amdgcn_wmma_f32_16x16x32_f16(
                    false, a0, false, bfrag, (short)0, acc[0][cb], false, false);
                acc[1][cb] = __builtin_amdgcn_wmma_f32_16x16x32_f16(
                    false, a1, false, bfrag, (short)0, acc[1][cb], false, false);
            }
        }
    }

    // ---- epilogue: D layout: row = tilebase + vv + half8, col = cb*16 + lane%16
#pragma unroll
    for (int t = 0; t < 2; ++t) {
        const int tbase = rowbase + t * 16;
        if (MODE == 2) {
#pragma unroll
            for (int vvi = 0; vvi < 8; ++vvi) {
                int row = tbase + vvi + half8;
                bool rv = row < rows;
                float x[8];
                float sum = 0.0f, sq = 0.0f;
#pragma unroll
                for (int cb = 0; cb < 8; ++cb) {
                    int col = cb * 16 + c0;
                    float v = acc[t][cb][vvi] + bias_s[col];
                    if (rv) v += res[(size_t)row * 128 + col];
                    x[cb] = v; sum += v; sq += v * v;
                }
                // reduce over the 16 lanes sharing this row (bit4 preserved)
#pragma unroll
                for (int off = 1; off < 16; off <<= 1) {
                    sum += __shfl_xor(sum, off, 32);
                    sq  += __shfl_xor(sq,  off, 32);
                }
                float mean = sum * (1.0f / 128.0f);
                float var  = sq * (1.0f / 128.0f) - mean * mean;
                float inv  = rsqrtf(var + 1e-5f);
                if (rv) {
#pragma unroll
                    for (int cb = 0; cb < 8; ++cb) {
                        int col = cb * 16 + c0;
                        out[(size_t)row * 128 + col] =
                            (x[cb] - mean) * inv * g_s[col] + b_s[col];
                    }
                }
            }
        } else {
#pragma unroll
            for (int vvi = 0; vvi < 8; ++vvi) {
                int row = tbase + vvi + half8;
                if (row >= rows) continue;
#pragma unroll
                for (int cb = 0; cb < 8; ++cb) {
                    int col = cb * 16 + c0;
                    float v = acc[t][cb][vvi] + bias_s[col];
                    if (MODE == 1) v = fmaxf(v, 0.0f);
                    out[(size_t)row * ldo + coloff + col] = v;
                }
            }
        }
    }
}

// ---------------------------------------------------------------------------
// Edge attention: score = K[src]*Q[dst]*pe/4 (written back over pe as e_attn),
// s = exp(clip(sum_d score,-5,5)); scatter wV[dst]+=V[src]*s, z[dst]+=s.
// 128 threads per edge (head = t/16, dim = t%16), 2 edges per block.
// ---------------------------------------------------------------------------
__global__ __launch_bounds__(256)
void edge_attn_kernel(const float* __restrict__ Q, const float* __restrict__ Kh,
                      const float* __restrict__ V, float* __restrict__ pe,
                      const int* __restrict__ src, const int* __restrict__ dst,
                      float* __restrict__ wV, float* __restrict__ z, int ne)
{
    int e = blockIdx.x * 2 + (threadIdx.x >> 7);
    if (e >= ne) return;
    int t = threadIdx.x & 127;
    int s = src[e], d = dst[e];
    float sc = Kh[(size_t)s * 128 + t] * Q[(size_t)d * 128 + t] * 0.25f
               * pe[(size_t)e * 128 + t];
    pe[(size_t)e * 128 + t] = sc;                 // e_attn (pre-exp), in place
    float hs = sc;
#pragma unroll
    for (int off = 1; off < 16; off <<= 1) hs += __shfl_xor(hs, off, 32);
    float sexp = __expf(fminf(fmaxf(hs, -5.0f), 5.0f));
    atomicAdd(&wV[(size_t)d * 128 + t], V[(size_t)s * 128 + t] * sexp);
    if ((t & 15) == 0) atomicAdd(&z[(size_t)d * 8 + (t >> 4)], sexp);
}

// h_attn = wV / (z + 1e-6)
__global__ void hattn_div(const float* __restrict__ wv, const float* __restrict__ z,
                          float* __restrict__ outp, int n)
{
    long i = (long)blockIdx.x * blockDim.x + threadIdx.x;
    if (i >= (long)n * 128) return;
    int node = (int)(i >> 7), t = (int)(i & 127);
    outp[i] = wv[i] / (z[(size_t)node * 8 + (t >> 4)] + 1e-6f);
}

__global__ void zero_kernel(float* __restrict__ p, long n)
{
    long i = (long)blockIdx.x * blockDim.x + threadIdx.x;
    if (i < n) p[i] = 0.0f;
}

// Transpose+convert weight: W[K][Nc] f32 -> Wt[Nc][K] f16
__global__ void wprep(const float* __restrict__ W, _Float16* __restrict__ Wt,
                      int Kd, int Nc)
{
    long i = (long)blockIdx.x * blockDim.x + threadIdx.x;
    if (i >= (long)Kd * Nc) return;
    int kk = (int)(i / Nc), col = (int)(i % Nc);
    Wt[(size_t)col * Kd + kk] = (_Float16)W[i];
}

// Padded input rows: [h_aa(32) | node_feat(64) | lap(8) | 0(24)]
__global__ void build_acat(const int* __restrict__ ref_aa, const int* __restrict__ alt_aa,
                           const int* __restrict__ gid, const float* __restrict__ aa_emb,
                           const float* __restrict__ node_feat, const float* __restrict__ lap,
                           float* __restrict__ acat, int n)
{
    long i = (long)blockIdx.x * blockDim.x + threadIdx.x;
    if (i >= (long)n * 128) return;
    int node = (int)(i >> 7), c = (int)(i & 127);
    float v = 0.0f;
    if (c < 32)       v = aa_emb[ref_aa[node] * 32 + c] * aa_emb[alt_aa[gid[node]] * 32 + c];
    else if (c < 96)  v = node_feat[(size_t)node * 64 + (c - 32)];
    else if (c < 104) v = lap[(size_t)node * 8 + (c - 96)];
    acat[i] = v;
}

// Fused padded input weight, transposed f16: rows 0..95 = nd.W, 96..103 = lap.W;
// bcat = nd.b + lap.b
__global__ void build_wcat(const float* __restrict__ ndW, const float* __restrict__ ndb,
                           const float* __restrict__ lapW, const float* __restrict__ lapb,
                           _Float16* __restrict__ wcatT, float* __restrict__ bcat)
{
    int idx = blockIdx.x * blockDim.x + threadIdx.x;
    if (idx < 128 * 128) {
        int kk = idx >> 7, col = idx & 127;
        float v = 0.0f;
        if (kk < 96)        v = ndW[kk * 128 + col];
        else if (kk < 104)  v = lapW[(kk - 96) * 128 + col];
        wcatT[col * 128 + kk] = (_Float16)v;
    }
    if (idx < 128) bcat[idx] = ndb[idx] + lapb[idx];
}

// Per-graph mean readout (graphs are contiguous blocks of NPG_ nodes)
__global__ void readout_mean(const float* __restrict__ h, float* __restrict__ hg)
{
    int g = blockIdx.x, c = threadIdx.x;  // 128 threads
    float acc = 0.0f;
    for (int n = g * NPG_; n < (g + 1) * NPG_; ++n) acc += h[(size_t)n * 128 + c];
    hg[(size_t)g * 128 + c] = acc * (1.0f / (float)NPG_);
}

// Head MLP: 128 -> 64 -> 32 (relu) -> 1, 8 graphs, one block
__global__ void mlp_head(const float* __restrict__ hg,
                         const float* __restrict__ W0, const float* __restrict__ b0,
                         const float* __restrict__ W1, const float* __restrict__ b1,
                         const float* __restrict__ W2, const float* __restrict__ b2,
                         float* __restrict__ outp)
{
    __shared__ float h_s[8 * 128];
    __shared__ float y1[8 * 64];
    __shared__ float y2[8 * 32];
    int tid = threadIdx.x;
    for (int i = tid; i < 8 * 128; i += 256) h_s[i] = hg[i];
    __syncthreads();
    for (int o = tid; o < 8 * 64; o += 256) {
        int b = o >> 6, j = o & 63;
        float a = b0[j];
        for (int c = 0; c < 128; ++c) a += h_s[b * 128 + c] * W0[c * 64 + j];
        y1[o] = fmaxf(a, 0.0f);
    }
    __syncthreads();
    for (int o = tid; o < 8 * 32; o += 256) {
        int b = o >> 5, j = o & 31;
        float a = b1[j];
        for (int c = 0; c < 64; ++c) a += y1[b * 64 + c] * W1[c * 32 + j];
        y2[o] = fmaxf(a, 0.0f);
    }
    __syncthreads();
    if (tid < 8) {
        float a = b2[0];
        for (int c = 0; c < 32; ++c) a += y2[tid * 32 + c] * W2[c];
        outp[tid] = a;
    }
}

// ---------------------------------------------------------------------------
extern "C" void kernel_launch(void* const* d_in, const int* in_sizes, int n_in,
                              void* d_out, int out_size, void* d_ws, size_t ws_size,
                              hipStream_t stream)
{
    (void)in_sizes; (void)n_in; (void)out_size; (void)ws_size;
    // setup_inputs() order
    const int*   ref_aa    = (const int*)d_in[0];
    const int*   alt_aa    = (const int*)d_in[1];
    const float* node_feat = (const float*)d_in[2];
    const float* edge_feat = (const float*)d_in[3];
    const float* lap       = (const float*)d_in[4];
    const int*   src       = (const int*)d_in[5];
    const int*   dst       = (const int*)d_in[6];
    const int*   gid       = (const int*)d_in[7];
    // params flattened with sorted dict keys (jax tree order)
    const float* aa_emb = (const float*)d_in[8];
    const float* edW    = (const float*)d_in[9];
    const float* edb    = (const float*)d_in[10];
    const float* lapW   = (const float*)d_in[11];
    const float* lapb   = (const float*)d_in[12];
    const float* mlpW0  = (const float*)d_in[61];
    const float* mlpb0  = (const float*)d_in[62];
    const float* mlpW1  = (const float*)d_in[63];
    const float* mlpb1  = (const float*)d_in[64];
    const float* mlpW2  = (const float*)d_in[65];
    const float* mlpb2  = (const float*)d_in[66];
    const float* ndW    = (const float*)d_in[67];
    const float* ndb    = (const float*)d_in[68];

    // ---- workspace carve-up: f32 activations, then f16 transposed weights
    float* ws = (float*)d_ws;
    size_t off = 0;
    float* hbuf = ws + off; off += (size_t)NODES * 128;
    float* qbuf = ws + off; off += (size_t)NODES * 128;   // acat / Q / h_attn / h1
    float* kbuf = ws + off; off += (size_t)NODES * 128;
    float* vbuf = ws + off; off += (size_t)NODES * 128;
    float* wvb  = ws + off; off += (size_t)NODES * 128;
    float* zb   = ws + off; off += (size_t)NODES * 8;
    float* ebuf = ws + off; off += (size_t)EDGES * 128;
    float* peb  = ws + off; off += (size_t)EDGES * 128;   // pe -> e_attn -> e1
    float* t256 = ws + off; off += (size_t)EDGES * 256;   // FFN hidden
    float* bcat = ws + off; off += 128;
    float* hg   = ws + off; off += BGR * 128;

    _Float16* hw = (_Float16*)(ws + off);
    size_t hoff = 0;
    _Float16* wcatT = hw + hoff; hoff += 128 * 128;
    _Float16* edWT  = hw + hoff; hoff += 32 * 128;
    _Float16 *WqT[2], *WkT[2], *WvT[2], *WeT[2], *OhT[2], *OeT[2];
    _Float16 *Fh1T[2], *Fh2T[2], *Fe1T[2], *Fe2T[2];
    for (int l = 0; l < 2; ++l) {
        Fe1T[l] = hw + hoff; hoff += 256 * 128;
        Fe2T[l] = hw + hoff; hoff += 128 * 256;
        Fh1T[l] = hw + hoff; hoff += 256 * 128;
        Fh2T[l] = hw + hoff; hoff += 128 * 256;
        OeT[l]  = hw + hoff; hoff += 128 * 128;
        OhT[l]  = hw + hoff; hoff += 128 * 128;
        WeT[l]  = hw + hoff; hoff += 128 * 128;
        WkT[l]  = hw + hoff; hoff += 128 * 128;
        WqT[l]  = hw + hoff; hoff += 128 * 128;
        WvT[l]  = hw + hoff; hoff += 128 * 128;
    }

    const int gN = (NODES + 255) / 256;   // 256 rows per block
    const int gE = (EDGES + 255) / 256;
    const long n128N = (long)NODES * 128;

    // ---- weight prep (transpose + f16) ----
    build_wcat<<<64, 256, 0, stream>>>(ndW, ndb, lapW, lapb, wcatT, bcat);
    wprep<<<(32 * 128 + 255) / 256, 256, 0, stream>>>(edW, edWT, 32, 128);
    for (int l = 0; l < 2; ++l) {
        int pb = 13 + 24 * l;
        wprep<<<128, 256, 0, stream>>>((const float*)d_in[pb + 0],  Fe1T[l], 128, 256);
        wprep<<<128, 256, 0, stream>>>((const float*)d_in[pb + 2],  Fe2T[l], 256, 128);
        wprep<<<128, 256, 0, stream>>>((const float*)d_in[pb + 4],  Fh1T[l], 128, 256);
        wprep<<<128, 256, 0, stream>>>((const float*)d_in[pb + 6],  Fh2T[l], 256, 128);
        wprep<<<64, 256, 0, stream>>>((const float*)d_in[pb + 8],   OeT[l],  128, 128);
        wprep<<<64, 256, 0, stream>>>((const float*)d_in[pb + 10],  OhT[l],  128, 128);
        wprep<<<64, 256, 0, stream>>>((const float*)d_in[pb + 12],  WeT[l],  128, 128);
        wprep<<<64, 256, 0, stream>>>((const float*)d_in[pb + 13],  WkT[l],  128, 128);
        wprep<<<64, 256, 0, stream>>>((const float*)d_in[pb + 14],  WqT[l],  128, 128);
        wprep<<<64, 256, 0, stream>>>((const float*)d_in[pb + 15],  WvT[l],  128, 128);
    }

    // ---- input projections ----
    build_acat<<<(unsigned)((n128N + 255) / 256), 256, 0, stream>>>(
        ref_aa, alt_aa, gid, aa_emb, node_feat, lap, qbuf, NODES);
    gemm_fused<128, 0><<<gN, 256, 0, stream>>>(qbuf, 128, wcatT, bcat,
        nullptr, nullptr, nullptr, hbuf, 128, 0, NODES);
    gemm_fused<32, 0><<<gE, 256, 0, stream>>>(edge_feat, 32, edWT, edb,
        nullptr, nullptr, nullptr, ebuf, 128, 0, EDGES);

    for (int l = 0; l < 2; ++l) {
        int pb = 13 + 24 * l;
        const float* Fe1b = (const float*)d_in[pb + 1];
        const float* Fe2b = (const float*)d_in[pb + 3];
        const float* Fh1b = (const float*)d_in[pb + 5];
        const float* Fh2b = (const float*)d_in[pb + 7];
        const float* Oeb  = (const float*)d_in[pb + 9];
        const float* Ohb  = (const float*)d_in[pb + 11];
        const float* l1eb = (const float*)d_in[pb + 16];
        const float* l1eg = (const float*)d_in[pb + 17];
        const float* l1hb = (const float*)d_in[pb + 18];
        const float* l1hg = (const float*)d_in[pb + 19];
        const float* l2eb = (const float*)d_in[pb + 20];
        const float* l2eg = (const float*)d_in[pb + 21];
        const float* l2hb = (const float*)d_in[pb + 22];
        const float* l2hg = (const float*)d_in[pb + 23];

        // QKV / pe projections (no bias)
        gemm_fused<128, 0><<<gN, 256, 0, stream>>>(hbuf, 128, WqT[l], nullptr,
            nullptr, nullptr, nullptr, qbuf, 128, 0, NODES);
        gemm_fused<128, 0><<<gN, 256, 0, stream>>>(hbuf, 128, WkT[l], nullptr,
            nullptr, nullptr, nullptr, kbuf, 128, 0, NODES);
        gemm_fused<128, 0><<<gN, 256, 0, stream>>>(hbuf, 128, WvT[l], nullptr,
            nullptr, nullptr, nullptr, vbuf, 128, 0, NODES);
        gemm_fused<128, 0><<<gE, 256, 0, stream>>>(ebuf, 128, WeT[l], nullptr,
            nullptr, nullptr, nullptr, peb, 128, 0, EDGES);

        // attention aggregation
        zero_kernel<<<(unsigned)((n128N + 255) / 256), 256, 0, stream>>>(wvb, n128N);
        zero_kernel<<<(unsigned)(((long)NODES * 8 + 255) / 256), 256, 0, stream>>>(
            zb, (long)NODES * 8);
        edge_attn_kernel<<<(EDGES + 1) / 2, 256, 0, stream>>>(
            qbuf, kbuf, vbuf, peb, src, dst, wvb, zb, EDGES);
        hattn_div<<<(unsigned)((n128N + 255) / 256), 256, 0, stream>>>(wvb, zb, qbuf, NODES);

        // node residual + LN + FFN
        gemm_fused<128, 2><<<gN, 256, 0, stream>>>(qbuf, 128, OhT[l], Ohb,
            hbuf, l1hg, l1hb, qbuf, 128, 0, NODES);                        // h1 in qbuf
        gemm_fused<128, 1><<<gN, 256, 0, stream>>>(qbuf, 128, Fh1T[l], Fh1b,
            nullptr, nullptr, nullptr, t256, 256, 0, NODES);
        gemm_fused<128, 1><<<gN, 256, 0, stream>>>(qbuf, 128, Fh1T[l] + 128 * 128,
            Fh1b + 128, nullptr, nullptr, nullptr, t256, 256, 128, NODES);
        gemm_fused<256, 2><<<gN, 256, 0, stream>>>(t256, 256, Fh2T[l], Fh2b,
            qbuf, l2hg, l2hb, hbuf, 128, 0, NODES);                        // h out

        // edge residual + LN + FFN
        gemm_fused<128, 2><<<gE, 256, 0, stream>>>(peb, 128, OeT[l], Oeb,
            ebuf, l1eg, l1eb, peb, 128, 0, EDGES);                         // e1 in peb
        gemm_fused<128, 1><<<gE, 256, 0, stream>>>(peb, 128, Fe1T[l], Fe1b,
            nullptr, nullptr, nullptr, t256, 256, 0, EDGES);
        gemm_fused<128, 1><<<gE, 256, 0, stream>>>(peb, 128, Fe1T[l] + 128 * 128,
            Fe1b + 128, nullptr, nullptr, nullptr, t256, 256, 128, EDGES);
        gemm_fused<256, 2><<<gE, 256, 0, stream>>>(t256, 256, Fe2T[l], Fe2b,
            peb, l2eg, l2eb, ebuf, 128, 0, EDGES);                         // e out
    }

    // readout + head
    readout_mean<<<BGR, 128, 0, stream>>>(hbuf, hg);
    mlp_head<<<1, 256, 0, stream>>>(hg, mlpW0, mlpb0, mlpW1, mlpb1, mlpW2, mlpb2,
                                    (float*)d_out);
}
